// GMMEstimationNetwork_22926535426510
// MI455X (gfx1250) — matrix operand order repaired and checked by reading, not compiled, not run on hardware
//
#include <hip/hip_runtime.h>

// ---------------- CDNA5 WMMA types ----------------
typedef __attribute__((ext_vector_type(16))) _Float16 v16h;
typedef __attribute__((ext_vector_type(8)))  _Float16 v8h;
typedef __attribute__((ext_vector_type(8)))  float    v8f;

#define NROWS   200000
#define DIM     64
#define H1DIM   256
#define H2DIM   128
#define KC      8
#define EPSV    1e-6f
#define LN_EPS  1e-5f

#define RPB     128          // rows per block (MLP kernel)
// LDS strides (in halves / floats); all f16 strides are multiples of 8 halves (16B)
#define ZR_S    68           // f32 z raw
#define XH_S    72           // f16 layernormed x
#define H1_S    264          // f16 hidden1
#define H2_S    136          // f16 hidden2
#define W1T_S   72           // W1 transposed [256][64+pad]
#define W2T_S   264          // W2 transposed [128][256+pad]
#define W3T_S   136          // W3 transposed [16][128+pad]

#define SIG_CHUNKS 128
#define SIG_ROWS   1563      // ceil(200000/128)
#define NST        136       // sigma kernel: d-major stride over 128 n's + pad

// ---- WMMA operand loaders (wave32; LDS row-major with given stride) ----
// A-matrix 16x32 f16: lane m = lane&15; K-base = (lane>=16)?8:0;
// half t -> K = base + t + (t>=8 ? 8 : 0)  => two contiguous 8-half (16B) reads.
__device__ __forceinline__ v16h ldsA(const _Float16* p, int m0, int k0, int stride) {
  int lane = threadIdx.x & 31;
  const _Float16* q = p + (m0 + (lane & 15)) * stride + k0 + ((lane >> 4) << 3);
  v8h lo = *(const v8h*)q;
  v8h hi = *(const v8h*)(q + 16);
  v16h a;
#pragma unroll
  for (int t = 0; t < 8; ++t) { a[t] = lo[t]; a[8 + t] = hi[t]; }
  return a;
}

// B-matrix 32x16 f16 from a TRANSPOSED buffer bt[col][k]:
// lane col = lane&15; K-base = (lane>=16)?8:0;
// half t -> K = base + t + (t>=8 ? 8 : 0)  => two contiguous 8-half (16B) reads.
__device__ __forceinline__ v16h ldsBT(const _Float16* p, int n0, int k0, int stride) {
  int lane = threadIdx.x & 31;
  const _Float16* q = p + (n0 + (lane & 15)) * stride + k0 + ((lane >> 4) << 3);
  v8h lo = *(const v8h*)q;
  v8h hi = *(const v8h*)(q + 16);
  v16h b;
#pragma unroll
  for (int t = 0; t < 8; ++t) { b[t] = lo[t]; b[8 + t] = hi[t]; }
  return b;
}

__device__ __forceinline__ v8f wmma_f16(v16h a, v16h b, v8f c) {
  return __builtin_amdgcn_wmma_f32_16x16x32_f16(false, a, false, b, (short)0, c, false, false);
}

// ---------------- kernel 0: zero workspace accumulators ----------------
__global__ void gmm_zero(float* __restrict__ p, int n) {
  int i = blockIdx.x * 256 + threadIdx.x;
  if (i < n) p[i] = 0.0f;
}

// ---------------- kernel 1: LayerNorm + MLP + softmax (gamma) ----------------
__global__ __launch_bounds__(256) void gmm_mlp_gamma(
    const float* __restrict__ z, const float* __restrict__ lnG, const float* __restrict__ lnB,
    const float* __restrict__ W1, const float* __restrict__ b1,
    const float* __restrict__ W2, const float* __restrict__ b2,
    const float* __restrict__ W3, const float* __restrict__ b3,
    float* __restrict__ gamma_out, float* __restrict__ ws_gs, float* __restrict__ ws_s)
{
  __shared__ __align__(16) _Float16 sW1t[H1DIM * W1T_S];     // 36.9 KB
  __shared__ __align__(16) _Float16 sW2t[H2DIM * W2T_S];     // 67.6 KB
  __shared__ __align__(16) _Float16 sW3t[16 * W3T_S];        //  4.4 KB
  __shared__ float sb1[H1DIM], sb2[H2DIM], sb3[KC], slnG[DIM], slnB[DIM];
  __shared__ __align__(16) float    zr[RPB * ZR_S];          // 34.8 KB (f32 z kept for moments)
  __shared__ __align__(16) _Float16 uXH2[RPB * H2_S];        // 34.8 KB union: xh early, h2 late
  __shared__ __align__(16) _Float16 h1[RPB * H1_S];          // 67.6 KB
  __shared__ float lg[RPB * KC];                             //  4.1 KB

  _Float16* xh = uXH2;   // stride XH_S, dead after GEMM1
  _Float16* h2 = uXH2;   // stride H2_S, written after GEMM2

  const int tid = threadIdx.x;

  // stage weights transposed -> f16 (global reads coalesced, LDS scatter once)
  for (int i = tid; i < DIM * H1DIM; i += 256) {
    int r = i >> 8, c = i & 255;
    sW1t[c * W1T_S + r] = (_Float16)W1[i];
  }
  for (int i = tid; i < H1DIM * H2DIM; i += 256) {
    int r = i >> 7, c = i & 127;
    sW2t[c * W2T_S + r] = (_Float16)W2[i];
  }
  for (int i = tid; i < H2DIM * 16; i += 256) {
    int r = i >> 4, c = i & 15;
    sW3t[c * W3T_S + r] = (_Float16)((c < KC) ? W3[r * KC + c] : 0.0f);
  }
  for (int i = tid; i < H1DIM; i += 256) sb1[i] = b1[i];
  for (int i = tid; i < H2DIM; i += 256) sb2[i] = b2[i];
  if (tid < KC) sb3[tid] = b3[tid];
  for (int i = tid; i < DIM; i += 256) { slnG[i] = lnG[i]; slnB[i] = lnB[i]; }

  const long rowBase = (long)blockIdx.x * RPB;
  for (int i = tid; i < RPB * DIM; i += 256) {
    int r = i >> 6, d = i & 63;
    long gr = rowBase + r;
    zr[r * ZR_S + d] = (gr < NROWS) ? z[gr * DIM + d] : 0.0f;
  }
  __syncthreads();

  // LayerNorm (fp32 VALU), emit f16 x
  if (tid < RPB) {
    const float* row = &zr[tid * ZR_S];
    float mean = 0.f;
#pragma unroll 8
    for (int d = 0; d < DIM; ++d) mean += row[d];
    mean *= (1.0f / DIM);
    float var = 0.f;
#pragma unroll 8
    for (int d = 0; d < DIM; ++d) { float dv = row[d] - mean; var += dv * dv; }
    var *= (1.0f / DIM);
    float rstd = rsqrtf(var + LN_EPS);
#pragma unroll 8
    for (int d = 0; d < DIM; ++d)
      xh[tid * XH_S + d] = (_Float16)(((row[d] - mean) * rstd) * slnG[d] + slnB[d]);
  }
  __syncthreads();

  const int w    = tid >> 5;
  const int m0   = w * 16;
  const int lane = tid & 31;
  const int ncol = lane & 15;
  const int mb   = (lane >> 4) << 3;

  // GEMM1: x[128x64] @ W1[64x256] -> relu -> h1 (f16)
  for (int ct = 0; ct < H1DIM / 16; ++ct) {
    v8f acc = {};
#pragma unroll
    for (int ks = 0; ks < 2; ++ks) {
      v16h a = ldsA(xh, m0, ks * 32, XH_S);
      v16h b = ldsBT(sW1t, ct * 16, ks * 32, W1T_S);
      acc = wmma_f16(a, b, acc);
    }
    float bias = sb1[ct * 16 + ncol];
#pragma unroll
    for (int v = 0; v < 8; ++v) {
      int r = m0 + v + mb;
      float h = fmaxf(acc[v] + bias, 0.0f);
      h1[r * H1_S + ct * 16 + ncol] = (_Float16)h;
    }
  }
  __syncthreads();

  // GEMM2: h1[128x256] @ W2[256x128] -> relu -> h2 (f16, overlays xh)
  for (int ct = 0; ct < H2DIM / 16; ++ct) {
    v8f acc = {};
#pragma unroll
    for (int ks = 0; ks < 8; ++ks) {
      v16h a = ldsA(h1, m0, ks * 32, H1_S);
      v16h b = ldsBT(sW2t, ct * 16, ks * 32, W2T_S);
      acc = wmma_f16(a, b, acc);
    }
    float bias = sb2[ct * 16 + ncol];
#pragma unroll
    for (int v = 0; v < 8; ++v) {
      int r = m0 + v + mb;
      float h = fmaxf(acc[v] + bias, 0.0f);
      h2[r * H2_S + ct * 16 + ncol] = (_Float16)h;
    }
  }
  __syncthreads();

  // GEMM3: h2[128x128] @ W3[128x8pad16] -> logits
  {
    v8f acc = {};
#pragma unroll
    for (int ks = 0; ks < 4; ++ks) {
      v16h a = ldsA(h2, m0, ks * 32, H2_S);
      v16h b = ldsBT(sW3t, 0, ks * 32, W3T_S);
      acc = wmma_f16(a, b, acc);
    }
    if (ncol < KC) {
#pragma unroll
      for (int v = 0; v < 8; ++v) lg[(m0 + v + mb) * KC + ncol] = acc[v];
    }
  }
  __syncthreads();

  // softmax (fp32), write gamma, keep gamma in LDS (zeroed for padded rows)
  if (tid < RPB) {
    long gr = rowBase + tid;
    float l[KC], mx = -3.4e38f;
#pragma unroll
    for (int j = 0; j < KC; ++j) { l[j] = lg[tid * KC + j] + sb3[j]; mx = fmaxf(mx, l[j]); }
    float s = 0.f;
#pragma unroll
    for (int j = 0; j < KC; ++j) { l[j] = __expf(l[j] - mx); s += l[j]; }
    float inv = 1.0f / s;
    bool valid = (gr < NROWS);
#pragma unroll
    for (int j = 0; j < KC; ++j) {
      float g = valid ? l[j] * inv : 0.0f;
      lg[tid * KC + j] = g;
      if (valid) gamma_out[gr * KC + j] = g;
    }
  }
  __syncthreads();

  // partial gamma_sum
  if (tid < KC) {
    float s = 0.f;
    for (int r = 0; r < RPB; ++r) s += lg[r * KC + tid];
    atomicAdd(&ws_gs[tid], s);
  }
  // partial s_c = sum gamma[n,c] * z[n,:]  (512 (c,d) pairs over 256 threads)
  for (int p = tid; p < KC * DIM; p += 256) {
    int c = p >> 6, d = p & 63;
    float s = 0.f;
    for (int r = 0; r < RPB; ++r) s += lg[r * KC + c] * zr[r * ZR_S + d];
    atomicAdd(&ws_s[p], s);
  }
}

// ---------------- kernel 2: phi, mu ----------------
__global__ __launch_bounds__(256) void gmm_finalize_phimu(
    const float* __restrict__ ws_gs, const float* __restrict__ ws_s,
    float* __restrict__ phi_out, float* __restrict__ mu_out, float* __restrict__ ws_mu)
{
  __shared__ float ph[KC];
  __shared__ float psum;
  int tid = threadIdx.x;
  if (tid < KC) ph[tid] = ws_gs[tid] * (1.0f / (float)NROWS) + EPSV / (float)KC;
  __syncthreads();
  if (tid == 0) { float s = 0.f; for (int c = 0; c < KC; ++c) s += ph[c]; psum = s; }
  __syncthreads();
  if (tid < KC) phi_out[tid] = ph[tid] / psum;
  for (int p = tid; p < KC * DIM; p += 256) {
    int c = p >> 6;
    float m = ws_s[p] / (ws_gs[c] + EPSV);
    mu_out[p] = m;
    ws_mu[p]  = m;
  }
}

// ---------------- kernel 3: S_c = Z^T diag(gamma_c) Z via WMMA ----------------
__global__ __launch_bounds__(256) void gmm_sigma_S(
    const float* __restrict__ z, const float* __restrict__ gamma, float* __restrict__ ws_S)
{
  __shared__ __align__(16) _Float16 zt[DIM * NST];   // d-major: zt[d][n]
  __shared__ __align__(16) _Float16 gzt[DIM * NST];  // d-major: gamma*z
  __shared__ float garr[128];

  const int c     = blockIdx.x & (KC - 1);
  const long row0 = (long)(blockIdx.x >> 3) * SIG_ROWS;
  const long row1 = (row0 + SIG_ROWS < (long)NROWS) ? row0 + SIG_ROWS : (long)NROWS;

  const int tid = threadIdx.x, lane = tid & 31, w = tid >> 5;
  const int i0 = (w >> 1) * 16;          // waves pair up on same i-tile (share A)
  const int j0 = ((w & 1) * 2) * 16;
  const int j1 = j0 + 16;

  v8f acc0 = {}, acc1 = {};

  for (long nb = row0; nb < row1; nb += 128) {
    __syncthreads();
    if (tid < 128) {
      long n = nb + tid;
      garr[tid] = (n < row1) ? gamma[n * KC + c] : 0.0f;
    }
    __syncthreads();
    for (int idx = tid; idx < 128 * DIM; idx += 256) {
      int r = idx >> 6, d = idx & 63;
      long n = nb + r;
      float zv = (n < row1) ? z[n * DIM + d] : 0.0f;   // zeroed pad: no NaN into WMMA
      zt[d * NST + r]  = (_Float16)zv;
      gzt[d * NST + r] = (_Float16)(zv * garr[r]);
    }
    __syncthreads();
#pragma unroll
    for (int ks = 0; ks < 4; ++ks) {
      v16h a  = ldsA(gzt, i0, ks * 32, NST);    // A[m][k] = gamma*z[n=k][i0+m]
      v16h b0 = ldsBT(zt, j0, ks * 32, NST);    // B[k][j] = z[n=k][j0+j]
      v16h b1 = ldsBT(zt, j1, ks * 32, NST);
      acc0 = wmma_f16(a, b0, acc0);
      acc1 = wmma_f16(a, b1, acc1);
    }
  }

  const int ncol = lane & 15;
  const int mb   = (lane >> 4) << 3;
#pragma unroll
  for (int v = 0; v < 8; ++v) {
    int m = i0 + v + mb;
    atomicAdd(&ws_S[c * (DIM * DIM) + m * DIM + j0 + ncol], acc0[v]);
    atomicAdd(&ws_S[c * (DIM * DIM) + m * DIM + j1 + ncol], acc1[v]);
  }
}

// ---------------- kernel 4: sigma ----------------
__global__ __launch_bounds__(256) void gmm_finalize_sigma(
    const float* __restrict__ ws_S, const float* __restrict__ ws_gs,
    const float* __restrict__ ws_s, const float* __restrict__ ws_mu,
    float* __restrict__ sig_out)
{
  int idx = blockIdx.x * 256 + threadIdx.x;
  if (idx >= KC * DIM * DIM) return;
  int c = idx >> 12, rem = idx & 4095, i = rem >> 6, j = rem & 63;
  float gs = ws_gs[c];
  float mi = ws_mu[c * DIM + i], mj = ws_mu[c * DIM + j];
  float si = ws_s[c * DIM + i],  sj = ws_s[c * DIM + j];
  // exact expansion of sum_n g (z-mu)(z-mu)^T
  float v = (ws_S[idx] - mi * sj - si * mj + gs * mi * mj) / (gs + EPSV);
  if (i == j) v += EPSV;
  sig_out[idx] = v;
}

// ---------------- launch ----------------
extern "C" void kernel_launch(void* const* d_in, const int* in_sizes, int n_in,
                              void* d_out, int out_size, void* d_ws, size_t ws_size,
                              hipStream_t stream)
{
  const float* z   = (const float*)d_in[0];
  const float* lnG = (const float*)d_in[1];
  const float* lnB = (const float*)d_in[2];
  const float* W1  = (const float*)d_in[3];
  const float* b1  = (const float*)d_in[4];
  const float* W2  = (const float*)d_in[5];
  const float* b2  = (const float*)d_in[6];
  const float* W3  = (const float*)d_in[7];
  const float* b3  = (const float*)d_in[8];

  float* out       = (float*)d_out;
  float* gamma_out = out;                            // [N,K]
  float* phi_out   = out + (long)NROWS * KC;         // [K]
  float* mu_out    = phi_out + KC;                   // [K,D]
  float* sig_out   = mu_out + KC * DIM;              // [K,D,D]

  float* ws    = (float*)d_ws;
  float* ws_gs = ws;                                 // 8
  float* ws_s  = ws + KC;                            // 512
  float* ws_S  = ws + KC + KC * DIM;                 // 32768
  float* ws_mu = ws_S + KC * DIM * DIM;              // 512

  const int nzero = KC + KC * DIM + KC * DIM * DIM;
  gmm_zero<<<(nzero + 255) / 256, 256, 0, stream>>>(ws, nzero);

  const int blocksA = (NROWS + RPB - 1) / RPB;       // 1563
  gmm_mlp_gamma<<<blocksA, 256, 0, stream>>>(z, lnG, lnB, W1, b1, W2, b2, W3, b3,
                                             gamma_out, ws_gs, ws_s);

  gmm_finalize_phimu<<<1, 256, 0, stream>>>(ws_gs, ws_s, phi_out, mu_out, ws_mu);

  gmm_sigma_S<<<SIG_CHUNKS * KC, 256, 0, stream>>>(z, gamma_out, ws_S);

  gmm_finalize_sigma<<<(KC * DIM * DIM + 255) / 256, 256, 0, stream>>>(
      ws_S, ws_gs, ws_s, ws_mu, sig_out);
}
